// LSTM_LSTM_455266533595
// MI455X (gfx1250) — matrix operand order repaired and checked by reading, not compiled, not run on hardware
//
#include <hip/hip_runtime.h>
#include <hip/hip_bf16.h>
#include <stdint.h>

// ---------------------------------------------------------------------------
// Seq2seq LSTM (2-layer encoder T=336, 2-layer decoder T=96), H=1024, B=256.
// FUSED per-step kernel: gate GEMM (v_wmma_f32_16x16x32_bf16, f32 accum)
// + LSTM nonlinearity + c/h state update in one pass. A wave owns the four
// gate n-tiles {nt, nt+64, nt+128, nt+192} of an (m-tile, d-tile) patch, so
// its accumulators are exactly the i/f/g/o gates of that patch. h is
// ping-pong buffered; c updates in place (owner-only).
// All global accesses use explicit addrspace(1) so loads lower to
// global_load_b128 (LOADcnt only), not flat_load (LOADcnt+DScnt).
// Weights fp32->bf16 once per launch (48MB, resident in 192MB L2).
// ---------------------------------------------------------------------------

typedef __attribute__((ext_vector_type(16))) __bf16    v16bf;
typedef __attribute__((ext_vector_type(8)))  float     v8f;
typedef __attribute__((ext_vector_type(4)))  unsigned int u32x4;  // trivial 16B

#define HID   1024
#define GH    4096      // 4*HID
#define BATCH 256
#define TSEQ  336
#define TOUT  96
#define KCHUNKS 32      // HID/32
#define MT 2            // m-tiles per wave (B-fragment reuse)

// ---- explicit global (addrspace 1) memory helpers -------------------------
__device__ __forceinline__ u32x4 gload_u4(const void* p) {
  return *(const __attribute__((address_space(1))) u32x4*)(uintptr_t)p;
}
__device__ __forceinline__ float gload_f(const float* p) {
  return *(const __attribute__((address_space(1))) float*)(uintptr_t)p;
}
__device__ __forceinline__ uint16_t gload_u16(const uint16_t* p) {
  return *(const __attribute__((address_space(1))) uint16_t*)(uintptr_t)p;
}
__device__ __forceinline__ void gstore_f(float* p, float v) {
  *(__attribute__((address_space(1))) float*)(uintptr_t)p = v;
}
__device__ __forceinline__ void gstore_u16(uint16_t* p, uint16_t v) {
  *(__attribute__((address_space(1))) uint16_t*)(uintptr_t)p = v;
}

__device__ __forceinline__ uint16_t f32_to_bf16_rne(float f) {
  uint32_t u = __float_as_uint(f);
  u += 0x7FFFu + ((u >> 16) & 1u);
  return (uint16_t)(u >> 16);
}
__device__ __forceinline__ float bf16_to_f32(uint16_t h) {
  return __uint_as_float(((uint32_t)h) << 16);
}
__device__ __forceinline__ float sigmoidf(float x) {
  return 1.0f / (1.0f + __expf(-x));
}

union Frag16 { v16bf v; u32x4 q[2]; };

// ---- fp32 -> bf16 weight conversion (one-shot, HBM streaming) -------------
__global__ void conv_bf16_kernel(const float* __restrict__ src,
                                 uint16_t* __restrict__ dst, int n) {
  int i = blockIdx.x * blockDim.x + threadIdx.x;
  if (i < n) dst[i] = f32_to_bf16_rne(gload_f(src + i));
}

// ---- one GEMM pass: acc[gate][mtile] += A @ Wgate^T -----------------------
// Fragment layouts used here (16-bit WMMA, wave32; contiguous k order):
//   A lane(g=lane>>4, m=lane&15): row m, k in [16g,16g+16)  (contiguous 32B)
//   B lane(g, n=lane&15):         row n, k in [16g,16g+16)  (contiguous 32B)
//   C/D lane(g, n): vgpr v holds element (m = v + 8g, n)
__device__ __forceinline__ void gemm_pair(
    v8f acc[4][MT],
    const uint16_t* __restrict__ Ap,   // bf16 [BATCH][HID]
    const uint16_t* __restrict__ Wp,   // bf16 [GH][HID], row = gate*HID + d
    int mg, int g, int ln, int d)
{
  const uint16_t* w0 = Wp + (size_t)(0 * HID + d) * HID + g * 16;
  const uint16_t* w1 = Wp + (size_t)(1 * HID + d) * HID + g * 16;
  const uint16_t* w2 = Wp + (size_t)(2 * HID + d) * HID + g * 16;
  const uint16_t* w3 = Wp + (size_t)(3 * HID + d) * HID + g * 16;
  for (int kc = 0; kc < KCHUNKS; ++kc) {
    Frag16 bf[4];
    bf[0].q[0] = gload_u4(w0 + kc * 32); bf[0].q[1] = gload_u4(w0 + kc * 32 + 8);
    bf[1].q[0] = gload_u4(w1 + kc * 32); bf[1].q[1] = gload_u4(w1 + kc * 32 + 8);
    bf[2].q[0] = gload_u4(w2 + kc * 32); bf[2].q[1] = gload_u4(w2 + kc * 32 + 8);
    bf[3].q[0] = gload_u4(w3 + kc * 32); bf[3].q[1] = gload_u4(w3 + kc * 32 + 8);
    #pragma unroll
    for (int i = 0; i < MT; ++i) {
      const uint16_t* ap =
          Ap + (size_t)((mg + i) * 16 + ln) * HID + kc * 32 + g * 16;
      Frag16 af;
      af.q[0] = gload_u4(ap);                   // k = kc*32 + 16g   .. +8
      af.q[1] = gload_u4(ap + 8);               // k = kc*32+16g+8  .. +8
      #pragma unroll
      for (int gate = 0; gate < 4; ++gate)
        acc[gate][i] = __builtin_amdgcn_wmma_f32_16x16x32_bf16(
            false, af.v, false, bf[gate].v, (short)0, acc[gate][i],
            false, false);
    }
  }
}

// ---- fused LSTM step ------------------------------------------------------
// gates = bias + A0@W0^T [+ A1@W1^T] ; then i/f/g/o -> (c,h) for the patch.
__global__ __launch_bounds__(128) void lstm_step_kernel(
    const uint16_t* __restrict__ A0, const uint16_t* __restrict__ W0,
    const uint16_t* __restrict__ A1, const uint16_t* __restrict__ W1,
    const float* __restrict__ bias,
    float* __restrict__ c, uint16_t* __restrict__ hout,
    const float* __restrict__ xs, int xoff, int xstride,
    const float* __restrict__ WihCol)
{
  const int lane = threadIdx.x & 31;
  const int wid  = blockIdx.x * 4 + (threadIdx.x >> 5);   // 0..511
  const int nh   = wid >> 3;                              // d-tile 0..63
  const int mg   = (wid & 7) * MT;                        // first m-tile
  const int g    = lane >> 4;
  const int ln   = lane & 15;
  const int d    = nh * 16 + ln;                          // this lane's d

  v8f acc[4][MT];                                         // [gate][mtile]
  #pragma unroll
  for (int gate = 0; gate < 4; ++gate) {
    float bv = gload_f(bias + gate * HID + d);
    #pragma unroll
    for (int i = 0; i < MT; ++i)
      #pragma unroll
      for (int v = 0; v < 8; ++v) acc[gate][i][v] = bv;
  }

  gemm_pair(acc, A0, W0, mg, g, ln, d);
  if (A1) gemm_pair(acc, A1, W1, mg, g, ln, d);           // uniform branch

  // ---- LSTM cell on the accumulators (EXEC stays all-ones) ----------------
  float wxi = 0.f, wxf = 0.f, wxg = 0.f, wxo = 0.f;
  if (WihCol) {                                           // input-dim-1 layers
    wxi = gload_f(WihCol + 0 * HID + d);
    wxf = gload_f(WihCol + 1 * HID + d);
    wxg = gload_f(WihCol + 2 * HID + d);
    wxo = gload_f(WihCol + 3 * HID + d);
  }
  #pragma unroll
  for (int i = 0; i < MT; ++i) {
    #pragma unroll
    for (int v = 0; v < 8; ++v) {
      int b = (mg + i) * 16 + 8 * g + v;                  // batch row (m=v+8g)
      float gi = acc[0][i][v];
      float gf = acc[1][i][v];
      float gg = acc[2][i][v];
      float go = acc[3][i][v];
      if (WihCol) {
        float xv = gload_f(xs + b * xstride + xoff);
        gi += xv * wxi; gf += xv * wxf; gg += xv * wxg; go += xv * wxo;
      }
      size_t idx = (size_t)b * HID + d;
      float cn = sigmoidf(gf) * gload_f(c + idx) + sigmoidf(gi) * tanhf(gg);
      gstore_f(c + idx, cn);
      gstore_u16(hout + idx, f32_to_bf16_rne(sigmoidf(go) * tanhf(cn)));
    }
  }
}

// ---- decoder head: one wave per batch row, wave32 butterfly reduce --------
__global__ void fc_head_kernel(const uint16_t* __restrict__ h1,
                               const float* __restrict__ fcW,
                               const float* __restrict__ fcb,
                               float* __restrict__ out,
                               float* __restrict__ xt, int t)
{
  int lane = threadIdx.x & 31;
  int b = blockIdx.x * (blockDim.x >> 5) + (threadIdx.x >> 5);
  const uint16_t* hp = h1 + (size_t)b * HID;
  float s = 0.f;
  for (int d = lane; d < HID; d += 32)
    s += bf16_to_f32(gload_u16(hp + d)) * gload_f(fcW + d);
  #pragma unroll
  for (int off = 16; off > 0; off >>= 1)
    s += __shfl_xor(s, off, 32);                          // wave32 reduce
  if (lane == 0) {
    float y = s + gload_f(fcb);
    gstore_f(out + b * TOUT + t, y);
    gstore_f(xt + b, y);
  }
}

__global__ void init_xt_kernel(const float* __restrict__ x,
                               float* __restrict__ xt) {
  int b = threadIdx.x;
  gstore_f(xt + b, gload_f(x + b * TSEQ + (TSEQ - 1)));   // x[:, -1, 0]
}

// ---------------------------------------------------------------------------
extern "C" void kernel_launch(void* const* d_in, const int* in_sizes, int n_in,
                              void* d_out, int out_size, void* d_ws, size_t ws_size,
                              hipStream_t stream) {
  const float* x     = (const float*)d_in[0];
  const float* eWih0 = (const float*)d_in[1];
  const float* eWhh0 = (const float*)d_in[2];
  const float* eb0   = (const float*)d_in[3];
  const float* eWih1 = (const float*)d_in[4];
  const float* eWhh1 = (const float*)d_in[5];
  const float* eb1   = (const float*)d_in[6];
  const float* dWih0 = (const float*)d_in[7];
  const float* dWhh0 = (const float*)d_in[8];
  const float* db0   = (const float*)d_in[9];
  const float* dWih1 = (const float*)d_in[10];
  const float* dWhh1 = (const float*)d_in[11];
  const float* db1   = (const float*)d_in[12];
  const float* fcW   = (const float*)d_in[13];
  const float* fcb   = (const float*)d_in[14];
  float* out = (float*)d_out;

  // Workspace carve-up (~52 MB)
  char* ws = (char*)d_ws;
  const size_t WN = (size_t)GH * HID;                     // 4M elems/matrix
  uint16_t* wEWhh0 = (uint16_t*)ws; ws += WN * 2;
  uint16_t* wEWih1 = (uint16_t*)ws; ws += WN * 2;
  uint16_t* wEWhh1 = (uint16_t*)ws; ws += WN * 2;
  uint16_t* wDWhh0 = (uint16_t*)ws; ws += WN * 2;
  uint16_t* wDWih1 = (uint16_t*)ws; ws += WN * 2;
  uint16_t* wDWhh1 = (uint16_t*)ws; ws += WN * 2;
  const size_t HN = (size_t)BATCH * HID;
  uint16_t* h0buf[2]; uint16_t* h1buf[2];
  h0buf[0] = (uint16_t*)ws; ws += HN * 2;
  h0buf[1] = (uint16_t*)ws; ws += HN * 2;
  h1buf[0] = (uint16_t*)ws; ws += HN * 2;
  h1buf[1] = (uint16_t*)ws; ws += HN * 2;
  float* c0 = (float*)ws;   ws += HN * 4;
  float* c1 = (float*)ws;   ws += HN * 4;
  float* xt = (float*)ws;   ws += BATCH * 4;

  // One-shot fp32 -> bf16 weight conversion (stays hot in 192MB L2)
  {
    int blk = 256, grd = (int)((WN + 255) / 256);
    conv_bf16_kernel<<<grd, blk, 0, stream>>>(eWhh0, wEWhh0, (int)WN);
    conv_bf16_kernel<<<grd, blk, 0, stream>>>(eWih1, wEWih1, (int)WN);
    conv_bf16_kernel<<<grd, blk, 0, stream>>>(eWhh1, wEWhh1, (int)WN);
    conv_bf16_kernel<<<grd, blk, 0, stream>>>(dWhh0, wDWhh0, (int)WN);
    conv_bf16_kernel<<<grd, blk, 0, stream>>>(dWih1, wDWih1, (int)WN);
    conv_bf16_kernel<<<grd, blk, 0, stream>>>(dWhh1, wDWhh1, (int)WN);
  }
  (void)hipMemsetAsync(h0buf[0], 0, HN * 2, stream);
  (void)hipMemsetAsync(h0buf[1], 0, HN * 2, stream);
  (void)hipMemsetAsync(h1buf[0], 0, HN * 2, stream);
  (void)hipMemsetAsync(h1buf[1], 0, HN * 2, stream);
  (void)hipMemsetAsync(c0, 0, HN * 4, stream);
  (void)hipMemsetAsync(c1, 0, HN * 4, stream);

  dim3 sGrid(128), sBlk(128);       // 512 waves: 64 d-tiles x 8 m-groups

  int cur = 0;
  // ---- encoder: layers interleaved so ys0 never materializes --------------
  for (int t = 0; t < TSEQ; ++t) {
    int nxt = cur ^ 1;
    // layer 0: gates = b0 + h0@Whh0^T + x_t*Wih0 (input dim 1)
    lstm_step_kernel<<<sGrid, sBlk, 0, stream>>>(
        h0buf[cur], wEWhh0, nullptr, nullptr, eb0,
        c0, h0buf[nxt], x, t, TSEQ, eWih0);
    // layer 1: gates = b1 + h0_new@Wih1^T + h1@Whh1^T (fused dual-GEMM)
    lstm_step_kernel<<<sGrid, sBlk, 0, stream>>>(
        h0buf[nxt], wEWih1, h1buf[cur], wEWhh1, eb1,
        c1, h1buf[nxt], nullptr, 0, 0, nullptr);
    cur = nxt;
  }

  // ---- decoder -------------------------------------------------------------
  init_xt_kernel<<<1, 256, 0, stream>>>(x, xt);
  for (int t = 0; t < TOUT; ++t) {
    int nxt = cur ^ 1;
    lstm_step_kernel<<<sGrid, sBlk, 0, stream>>>(
        h0buf[cur], wDWhh0, nullptr, nullptr, db0,
        c0, h0buf[nxt], xt, 0, 1, dWih0);
    lstm_step_kernel<<<sGrid, sBlk, 0, stream>>>(
        h0buf[nxt], wDWih1, h1buf[cur], wDWhh1, db1,
        c1, h1buf[nxt], nullptr, 0, 0, nullptr);
    fc_head_kernel<<<32, 256, 0, stream>>>(h1buf[nxt], fcW, fcb, out, xt, t);
    cur = nxt;
  }
}
